// APF_36120674959459
// MI455X (gfx1250) — matrix operand order repaired and checked by reading, not compiled, not run on hardware
//
#include <hip/hip_runtime.h>
#include <hip/hip_bf16.h>

// Problem constants from the reference
#define BB   2
#define GG   512     // coarse points M
#define NN   4096    // dense queries
#define CC   768     // feature channels
#define KK   4       // knn k
#define GROUPS 4
#define EPS  1e-5f
#define SLOPE 0.2f

typedef __attribute__((ext_vector_type(16))) _Float16 v16h;
typedef __attribute__((ext_vector_type(8)))  float    v8f;

// ---------------------------------------------------------------------------
// KNN: for each query n, find k=4 nearest of M reference points.
// d = |q|^2 + |r|^2 - 2 q.r  (same form as reference). Strict-< insertion
// keeps the earliest index on ties, matching jax.lax.top_k behavior.
// ---------------------------------------------------------------------------
__global__ void knn_kernel(const float* __restrict__ ref,
                           const float* __restrict__ qry,
                           int* __restrict__ idx,
                           int M, int N) {
    const int b = blockIdx.y;
    const int n = blockIdx.x * blockDim.x + threadIdx.x;
    const float* rb = ref + (size_t)b * 3 * M;
    const float* qb = qry + (size_t)b * 3 * N;

    float qx = 0.f, qy = 0.f, qz = 0.f, qq = 0.f;
    if (n < N) {
        qx = qb[n]; qy = qb[N + n]; qz = qb[2 * N + n];
        qq = qx * qx + qy * qy + qz * qz;
    }

    float d0 = 3.4e38f, d1 = 3.4e38f, d2 = 3.4e38f, d3 = 3.4e38f;
    int   i0 = 0, i1 = 0, i2 = 0, i3 = 0;

    __shared__ float sx[1024], sy[1024], sz[1024], sw[1024];

    for (int m0 = 0; m0 < M; m0 += 1024) {
        int cnt = M - m0; if (cnt > 1024) cnt = 1024;
        for (int t = threadIdx.x; t < cnt; t += blockDim.x) {
            float rx = rb[m0 + t];
            float ry = rb[M + m0 + t];
            float rz = rb[2 * M + m0 + t];
            sx[t] = rx; sy[t] = ry; sz[t] = rz;
            sw[t] = rx * rx + ry * ry + rz * rz;
        }
        __syncthreads();
        if (n < N) {
            for (int j = 0; j < cnt; ++j) {
                float d = qq + sw[j] - 2.0f * (qx * sx[j] + qy * sy[j] + qz * sz[j]);
                int m = m0 + j;
                if (d < d3) {
                    if (d < d2) {
                        d3 = d2; i3 = i2;
                        if (d < d1) {
                            d2 = d1; i2 = i1;
                            if (d < d0) { d1 = d0; i1 = i0; d0 = d; i0 = m; }
                            else        { d1 = d;  i1 = m; }
                        } else { d2 = d; i2 = m; }
                    } else { d3 = d; i3 = m; }
                }
            }
        }
        __syncthreads();
    }

    if (n < N) {
        int* o = idx + ((size_t)b * N + n) * KK;
        o[0] = i0; o[1] = i1; o[2] = i2; o[3] = i3;
    }
}

// ---------------------------------------------------------------------------
// Fused edge-gather + GEMM using v_wmma_f32_16x16x32_f16.
//   y[b][o][col] = sum_c W[o][c] * E[c][col],  col = n*4+kk, Cin = 2*Chalf
//   E[c][col] = c < Chalf ? fk[b][c][g] - fq[b][c][n] : fq[b][c-Chalf][n]
//
// Block: 256 threads = 8 waves; block tile = 128 output rows x 64 columns.
// Each wave owns a 16-row slice and computes 4 adjacent 16x16 D tiles,
// reusing one A (weight) fragment against 4 B (edge) fragments -> 4 WMMAs
// per 32-wide K step per wave.
//
// LDS tiles are stored PRE-SWIZZLED in the per-lane fragment layout so each
// lane fetches its entire 16-half fragment as one aligned 32B read
// (ds_load_b128 x2), with no unpacking VALU.
//   A 16x32 f16: element (m,K) -> lane = m + 16*bit3(K), e = (K&7)+8*(K>=16)
//   B 32x16 f16: element (K,n) -> lane = n + 16*bit3(K), e = (K&7)+8*(K>=16)
// ---------------------------------------------------------------------------
__global__ void edge_wmma_kernel(const float* __restrict__ fk,
                                 const float* __restrict__ fq,
                                 const float* __restrict__ W,
                                 const int*   __restrict__ idx,
                                 float*       __restrict__ y,
                                 int Chalf, int Mkeys, int Cout, int N) {
    const int Cin     = 2 * Chalf;
    const int b       = blockIdx.z;
    const int colBase = blockIdx.x * 64;   // over N*KK columns
    const int oBlock  = blockIdx.y * 128;  // 8 waves * 16 rows
    const int tid     = threadIdx.x;
    const int lane    = tid & 31;
    const int wave    = tid >> 5;

    __shared__ int  sN[64];
    __shared__ int  sG[64];
    __shared__ v16h aBuf[8][32];   // [wave-slice][lane] : pre-swizzled A frags
    __shared__ v16h bBuf[4][32];   // [col-tile][lane]   : pre-swizzled B frags

    if (tid < 64) {
        int col = colBase + tid;
        int n = col >> 2, kk = col & 3;
        sN[tid] = n;
        sG[tid] = idx[((size_t)b * N + n) * KK + kk];
    }

    const float* fkb = fk + (size_t)b * Chalf * Mkeys;
    const float* fqb = fq + (size_t)b * Chalf * N;

    _Float16* ap = (_Float16*)&aBuf[0][0];
    _Float16* bp = (_Float16*)&bBuf[0][0];

    v8f acc0 = {}, acc1 = {}, acc2 = {}, acc3 = {};

    for (int kb = 0; kb < Cin; kb += 32) {
        __syncthreads();  // protects sN/sG (first iter) and prior-iter tiles

        // Stage gathered edge tile (32 K x 64 cols = 2048 elems, 8/thread),
        // writing straight into the B fragment layout.
        for (int e2 = tid; e2 < 2048; e2 += 256) {
            int cl = e2 >> 6;        // local K (0..31)
            int j  = e2 & 63;        // column (0..63)
            int c  = kb + cl;
            float v;
            if (c < Chalf) v = fkb[(size_t)c * Mkeys + sG[j]] - fqb[(size_t)c * N + sN[j]];
            else           v = fqb[(size_t)(c - Chalf) * N + sN[j]];
            int ct    = j >> 4;                      // which 16-col tile
            int blane = (j & 15) + ((cl & 8) ? 16 : 0);
            int be    = (cl & 7) + ((cl >= 16) ? 8 : 0);
            bp[((ct * 32 + blane) << 4) + be] = (_Float16)v;
        }
        // Stage weight tile (128 rows x 32 K = 4096 elems, 16/thread),
        // writing straight into the A fragment layout.
        for (int e2 = tid; e2 < 4096; e2 += 256) {
            int r  = e2 >> 5;        // local row (0..127)
            int cl = e2 & 31;        // local K  (0..31)
            float v = W[(size_t)(oBlock + r) * Cin + kb + cl];
            int ws    = r >> 4;                      // wave slice
            int alane = (r & 15) + ((cl & 8) ? 16 : 0);
            int ae    = (cl & 7) + ((cl >= 16) ? 8 : 0);
            ap[((ws * 32 + alane) << 4) + ae] = (_Float16)v;
        }
        __syncthreads();

        // Whole-fragment LDS reads: 32B per lane per fragment.
        v16h a  = aBuf[wave][lane];
        v16h b0 = bBuf[0][lane];
        v16h b1 = bBuf[1][lane];
        v16h b2 = bBuf[2][lane];
        v16h b3 = bBuf[3][lane];

        acc0 = __builtin_amdgcn_wmma_f32_16x16x32_f16(false, a, false, b0,
                                                      (short)0, acc0, false, false);
        acc1 = __builtin_amdgcn_wmma_f32_16x16x32_f16(false, a, false, b1,
                                                      (short)0, acc1, false, false);
        acc2 = __builtin_amdgcn_wmma_f32_16x16x32_f16(false, a, false, b2,
                                                      (short)0, acc2, false, false);
        acc3 = __builtin_amdgcn_wmma_f32_16x16x32_f16(false, a, false, b3,
                                                      (short)0, acc3, false, false);
    }

    // D layout: lanes 0-15 -> M=v, lanes 16-31 -> M=8+v; N = lane%16.
    const size_t cols = (size_t)N * KK;
    const int ncol = lane & 15;
    const int mofs = (lane >= 16) ? 8 : 0;
#pragma unroll
    for (int v = 0; v < 8; ++v) {
        int o = oBlock + wave * 16 + mofs + v;
        float* row = y + ((size_t)b * Cout + o) * cols + colBase + ncol;
        row[0]  = acc0[v];
        row[16] = acc1[v];
        row[32] = acc2[v];
        row[48] = acc3[v];
    }
}

// ---------------------------------------------------------------------------
// Per (batch, group) mean / inv-std over (Cout/GROUPS) * N * KK elements.
// One block per (b,g); fixed-order tree reduction -> deterministic.
// mv[(b*GROUPS+g)*2 + {0,1}] = {mean, invstd}
// ---------------------------------------------------------------------------
__global__ void stats_kernel(const float* __restrict__ y,
                             float* __restrict__ mv,
                             int Cout, size_t cols) {
    const int bg = blockIdx.x;
    const int b = bg / GROUPS, g = bg % GROUPS;
    const int cg = Cout / GROUPS;
    const float* base = y + ((size_t)b * Cout + (size_t)g * cg) * cols;
    const size_t total = (size_t)cg * cols;

    float s1 = 0.f, s2 = 0.f;
    for (size_t i = threadIdx.x; i < total; i += blockDim.x) {
        float v = base[i];
        s1 += v; s2 += v * v;
    }
    __shared__ float r1[256], r2[256];
    r1[threadIdx.x] = s1; r2[threadIdx.x] = s2;
    __syncthreads();
    for (int s = 128; s > 0; s >>= 1) {
        if (threadIdx.x < s) {
            r1[threadIdx.x] += r1[threadIdx.x + s];
            r2[threadIdx.x] += r2[threadIdx.x + s];
        }
        __syncthreads();
    }
    if (threadIdx.x == 0) {
        float inv = 1.0f / (float)total;
        float mean = r1[0] * inv;
        float var  = r2[0] * inv - mean * mean;
        mv[bg * 2 + 0] = mean;
        mv[bg * 2 + 1] = rsqrtf(var + EPS);
    }
}

// ---------------------------------------------------------------------------
// GroupNorm affine + LeakyReLU + max over k. One thread per (b,o,n).
// ---------------------------------------------------------------------------
__global__ void finalize_kernel(const float* __restrict__ y,
                                const float* __restrict__ mv,
                                const float* __restrict__ gamma,
                                const float* __restrict__ beta,
                                float* __restrict__ outp,
                                int Cout, int N) {
    size_t t = (size_t)blockIdx.x * blockDim.x + threadIdx.x;
    size_t total = (size_t)BB * Cout * N;
    if (t >= total) return;
    int b = (int)(t / ((size_t)Cout * N));
    size_t rem = t - (size_t)b * Cout * N;
    int o = (int)(rem / N);
    int n = (int)(rem % N);
    int g = o / (Cout / GROUPS);

    float mean = mv[(b * GROUPS + g) * 2 + 0];
    float istd = mv[(b * GROUPS + g) * 2 + 1];
    float ga = gamma[o], be = beta[o];

    const size_t cols = (size_t)N * KK;
    const float* yy = y + ((size_t)b * Cout + o) * cols + (size_t)n * KK;
    float mx = -3.4e38f;
#pragma unroll
    for (int kk = 0; kk < KK; ++kk) {
        float v = (yy[kk] - mean) * istd * ga + be;
        v = (v >= 0.f) ? v : SLOPE * v;
        mx = fmaxf(mx, v);
    }
    outp[((size_t)b * Cout + o) * N + n] = mx;
}

// ---------------------------------------------------------------------------
// Host-side launch
// ---------------------------------------------------------------------------
static inline size_t align256(size_t x) { return (x + 255) & ~(size_t)255; }

extern "C" void kernel_launch(void* const* d_in, const int* in_sizes, int n_in,
                              void* d_out, int out_size, void* d_ws, size_t ws_size,
                              hipStream_t stream) {
    (void)in_sizes; (void)n_in; (void)out_size; (void)ws_size;
    const float* coor   = (const float*)d_in[0];  // (B,3,G)
    const float* f      = (const float*)d_in[1];  // (B,C,G)
    const float* coor_q = (const float*)d_in[2];  // (B,3,N)
    const float* f_q    = (const float*)d_in[3];  // (B,C,N)
    const float* W1     = (const float*)d_in[4];  // (512,1536)
    const float* g1     = (const float*)d_in[5];
    const float* b1     = (const float*)d_in[6];
    const float* W2     = (const float*)d_in[7];  // (768,1024)
    const float* g2     = (const float*)d_in[8];
    const float* b2     = (const float*)d_in[9];
    float* out          = (float*)d_out;          // (B,768,N)

    // Workspace layout
    char* ws = (char*)d_ws;
    size_t off = 0;
    int*   idx1 = (int*)(ws + off);   off = align256(off + (size_t)BB * NN * KK * 4);
    int*   idx2 = (int*)(ws + off);   off = align256(off + (size_t)BB * NN * KK * 4);
    float* mv1  = (float*)(ws + off); off = align256(off + (size_t)BB * GROUPS * 2 * 4);
    float* mv2  = (float*)(ws + off); off = align256(off + (size_t)BB * GROUPS * 2 * 4);
    float* h    = (float*)(ws + off); off = align256(off + (size_t)BB * 512 * NN * 4);
    float* y1   = (float*)(ws + off); off = align256(off + (size_t)BB * 512 * NN * KK * 4);
    float* y2   = (float*)(ws + off); off = align256(off + (size_t)BB * 768 * NN * KK * 4);

    const size_t cols = (size_t)NN * KK;  // 16384

    // Stage 1: knn(coor -> coor_q), fused gather+GEMM, GN stats, finalize->h
    knn_kernel<<<dim3(NN / 256, BB), 256, 0, stream>>>(coor, coor_q, idx1, GG, NN);
    edge_wmma_kernel<<<dim3((unsigned)(cols / 64), 512 / 128, BB), 256, 0, stream>>>(
        f, f_q, W1, idx1, y1, /*Chalf=*/CC, /*Mkeys=*/GG, /*Cout=*/512, NN);
    stats_kernel<<<BB * GROUPS, 256, 0, stream>>>(y1, mv1, 512, cols);
    finalize_kernel<<<(unsigned)(((size_t)BB * 512 * NN + 255) / 256), 256, 0, stream>>>(
        y1, mv1, g1, b1, h, 512, NN);

    // Stage 2: self-knn on coor_q, edge conv over h, finalize -> out
    knn_kernel<<<dim3(NN / 256, BB), 256, 0, stream>>>(coor_q, coor_q, idx2, NN, NN);
    edge_wmma_kernel<<<dim3((unsigned)(cols / 64), 768 / 128, BB), 256, 0, stream>>>(
        h, h, W2, idx2, y2, /*Chalf=*/512, /*Mkeys=*/NN, /*Cout=*/768, NN);
    stats_kernel<<<BB * GROUPS, 256, 0, stream>>>(y2, mv2, 768, cols);
    finalize_kernel<<<(unsigned)(((size_t)BB * 768 * NN + 255) / 256), 256, 0, stream>>>(
        y2, mv2, g2, b2, out, 768, NN);
}
